// RSNN_79147657330801
// MI455X (gfx1250) — matrix operand order repaired and checked by reading, not compiled, not run on hardware
//
#include <hip/hip_runtime.h>

#define Bb   128
#define Tt   512
#define INn  128
#define NN   512
#define OUTo 128
#define BM   16
#define THREADS 512            // 16 wave32s

#define NA   (NN + 8)          // padded LDS row stride (halves): 1040B -> conflict-free
#define NXA  (INn + 8)

typedef __attribute__((ext_vector_type(16))) __bf16 v16bf;
typedef __attribute__((ext_vector_type(8)))  __bf16 v8bf;
typedef __attribute__((ext_vector_type(8)))  float  v8f;

static __device__ __forceinline__ unsigned short f2bf(float f) {
  union { float f; unsigned u; } x; x.f = f;
  unsigned u = x.u;
  return (unsigned short)((u + 0x7FFFu + ((u >> 16) & 1u)) >> 16);
}

// ---- weight pre-pack: B-matrix (K x Ncols) -> WMMA-B tile layout.
// Block (nt, kt) = 32x16 bf16; lane L = 16*((k>>3)&1) + n%16 owns 16 contiguous
// bf16 (j = (k&7) + 8*((k>>4)&1)) => one contiguous 32B read per lane per tile.

__global__ void pack_J(const float* __restrict__ J, unsigned short* __restrict__ JB) {
  int tid = blockIdx.x * blockDim.x + threadIdx.x;          // 512*512
  int n = tid >> 9, k = tid & 511;                           // B[k][n] = J[n][k]
  int nt = n >> 4, nl = n & 15;
  int kt = k >> 5, half = (k >> 3) & 1, j = (k & 7) + 8 * ((k >> 4) & 1);
  JB[((nt * 16 + kt) * 512) + (half * 16 + nl) * 16 + j] = f2bf(J[n * 512 + k]);
}

__global__ void pack_Bw(const float* __restrict__ Bw, unsigned short* __restrict__ BwB) {
  int tid = blockIdx.x * blockDim.x + threadIdx.x;          // 512*128
  int n = tid >> 7, i = tid & 127;                           // B[k=i][n] = Bw[n][i]
  int nt = n >> 4, nl = n & 15;
  int kt = i >> 5, half = (i >> 3) & 1, j = (i & 7) + 8 * ((i >> 4) & 1);
  BwB[((nt * 4 + kt) * 512) + (half * 16 + nl) * 16 + j] = f2bf(Bw[n * 128 + i]);
}

__global__ void pack_W(const float* __restrict__ W, unsigned short* __restrict__ WB) {
  int tid = blockIdx.x * blockDim.x + threadIdx.x;          // 128*512
  int o = tid >> 9, n = tid & 511;                           // B[k=n][o] = W[o][n]
  int ot = o >> 4, ol = o & 15;
  int kt = n >> 5, half = (n >> 3) & 1, j = (n & 7) + 8 * ((n >> 4) & 1);
  WB[((ot * 16 + kt) * 512) + (half * 16 + ol) * 16 + j] = f2bf(W[o * 512 + n]);
}

static __device__ __forceinline__ v8f wmma_bf16(v16bf a, v16bf b, v8f c) {
  return __builtin_amdgcn_wmma_f32_16x16x32_bf16(false, a, false, b,
                                                 (short)0, c, false, false);
}

// Phase C: out[:, tc, ot*16..+16] = lamb @ W^T for one OUT tile per wave
static __device__ __forceinline__ void phase_c(
    const unsigned short* WBt, const unsigned short* lambA_s,
    float* __restrict__ outseq, int b0, int tc,
    int wave, int lane, int half, int nl)
{
  const int ot = wave;
  v8f acc = {};
  const unsigned short* wb = WBt + (ot * 16) * 512 + lane * 16;
  const unsigned short* lrow = lambA_s + nl * NA + half * 8;
  #pragma unroll
  for (int kt = 0; kt < 16; ++kt) {
    union { v16bf v; v8bf h[2]; } a;
    a.h[0] = *(const v8bf*)(lrow + kt * 32);
    a.h[1] = *(const v8bf*)(lrow + kt * 32 + 16);
    v16bf b = *(const v16bf*)(wb + kt * 512);
    acc = wmma_bf16(a.v, b, acc);
  }
  const int o = ot * 16 + nl;
  #pragma unroll
  for (int r = 0; r < 8; ++r) {
    const int m = r + 8 * half;
    __builtin_nontemporal_store(acc[r],
        &outseq[(long)(b0 + m) * Tt * OUTo + (long)tc * OUTo + o]);
  }
}

// ---- main recurrent kernel: one workgroup owns BM=16 batch rows for all T.
// Per step: [C(t-1) on waves 0-7 || A(t) on waves 8-15] ; barrier ; B(t) ; barrier
__global__ __launch_bounds__(THREADS)
void rsnn_kernel(const float* __restrict__ inputs,
                 const float* __restrict__ rand_u,
                 const unsigned short* __restrict__ JB,
                 const unsigned short* __restrict__ BwB,
                 const unsigned short* __restrict__ WB,
                 float* __restrict__ vt,
                 float* __restrict__ zt,
                 float* __restrict__ outseq)
{
  __shared__ float vbuf[BM * NN];                             // 32 KB
  __shared__ float zbuf[BM * NN];                             // 32 KB
  __shared__ __align__(16) unsigned short tanhvA[BM * NA];    // 16.25 KB (padded)
  __shared__ __align__(16) unsigned short lambA[BM * NA];     // 16.25 KB (padded)
  __shared__ __align__(16) unsigned short xtA[BM * NXA];      //  4.25 KB (padded)

  const int tid  = threadIdx.x;
  const int wave = tid >> 5;
  const int lane = tid & 31;
  const int half = lane >> 4;
  const int nl   = lane & 15;
  const int b0   = blockIdx.x * BM;

  for (int idx = tid; idx < BM * NN; idx += THREADS) {
    vbuf[idx] = 0.f; zbuf[idx] = 0.f;
    int m = idx >> 9, n = idx & (NN - 1);
    long row = (long)(b0 + m) * (Tt + 1) * NN + n;
    __builtin_nontemporal_store(0.f, &vt[row]);
    __builtin_nontemporal_store(0.f, &zt[row]);
  }
  __syncthreads();

  for (int t = 0; t < Tt; ++t) {
    // LICM-killer on a zero SGPR *offset*: loads stay in-loop, but pointer
    // provenance (kernel arg) is preserved -> global_load, not flat_load.
    unsigned woff = 0;
    asm volatile("" : "+s"(woff));
    const unsigned short* JBt  = JB  + woff;
    const unsigned short* BwBt = BwB + woff;
    const unsigned short* WBt  = WB  + woff;

    // ---- merged phase: C(t-1) on waves 0-7, A(t) on waves 8-15 ----------
    if (wave >= 8) {
      const int atid = tid - 256;
      for (int idx = atid; idx < BM * NN; idx += 256) {
        float vn = 0.9f * vbuf[idx] + 0.1f * zbuf[idx];
        vbuf[idx] = vn;
        int m = idx >> 9, n = idx & (NN - 1);
        __builtin_nontemporal_store(vn,
            &vt[(long)(b0 + m) * (Tt + 1) * NN + (long)(t + 1) * NN + n]);
        tanhvA[m * NA + n] = f2bf(tanhf(vn));
      }
      for (int idx = atid; idx < BM * INn; idx += 256) {
        int m = idx >> 7, i = idx & (INn - 1);
        xtA[m * NXA + i] = f2bf(__builtin_nontemporal_load(
            &inputs[(long)(b0 + m) * Tt * INn + (long)t * INn + i]));
      }
    } else if (t > 0) {
      phase_c(WBt, lambA, outseq, b0, t - 1, wave, lane, half, nl);
    }
    __syncthreads();

    // ---- Phase B: lamb = tanh(v)@J^T + x@Bw^T; two 16-col tiles per wave
    {
      const int nt0 = wave * 2;
      v8f acc0 = {}, acc1 = {};
      const unsigned short* jb0 = JBt + (nt0 * 16) * 512 + lane * 16;
      const unsigned short* jb1 = JBt + ((nt0 + 1) * 16) * 512 + lane * 16;
      const unsigned short* arow = tanhvA + nl * NA + half * 8;
      #pragma unroll
      for (int kt = 0; kt < 16; ++kt) {
        union { v16bf v; v8bf h[2]; } a;
        a.h[0] = *(const v8bf*)(arow + kt * 32);
        a.h[1] = *(const v8bf*)(arow + kt * 32 + 16);
        v16bf b0 = *(const v16bf*)(jb0 + kt * 512);
        v16bf b1 = *(const v16bf*)(jb1 + kt * 512);
        acc0 = wmma_bf16(a.v, b0, acc0);
        acc1 = wmma_bf16(a.v, b1, acc1);
      }
      const unsigned short* bw0 = BwBt + (nt0 * 4) * 512 + lane * 16;
      const unsigned short* bw1 = BwBt + ((nt0 + 1) * 4) * 512 + lane * 16;
      const unsigned short* xrow = xtA + nl * NXA + half * 8;
      #pragma unroll
      for (int kt = 0; kt < 4; ++kt) {
        union { v16bf v; v8bf h[2]; } a;
        a.h[0] = *(const v8bf*)(xrow + kt * 32);
        a.h[1] = *(const v8bf*)(xrow + kt * 32 + 16);
        v16bf b0 = *(const v16bf*)(bw0 + kt * 512);
        v16bf b1 = *(const v16bf*)(bw1 + kt * 512);
        acc0 = wmma_bf16(a.v, b0, acc0);
        acc1 = wmma_bf16(a.v, b1, acc1);
      }
      // spike epilogue for both tiles
      #pragma unroll
      for (int tt = 0; tt < 2; ++tt) {
        const int n = (nt0 + tt) * 16 + nl;
        const v8f acc = tt ? acc1 : acc0;
        #pragma unroll
        for (int r = 0; r < 8; ++r) {
          const int m = r + 8 * half;                        // C/D layout
          float lamb = acc[r];
          float u = __builtin_nontemporal_load(
              &rand_u[(long)(b0 + m) * Tt * NN + (long)t * NN + n]);
          float s = (lamb - 0.4f) * 2.5f;                    // (lamb-THR)/THR
          float sig = 1.0f / (1.0f + __expf(-0.1f * s));     // sigmoid(TEMP*s)
          float znew = sig > u ? 1.0f : 0.0f;
          zbuf[m * NN + n] = znew;
          __builtin_nontemporal_store(znew,
              &zt[(long)(b0 + m) * (Tt + 1) * NN + (long)(t + 1) * NN + n]);
          lambA[m * NA + n] = f2bf(lamb);
        }
      }
    }
    __syncthreads();
  }

  // ---- drain: C for the last step --------------------------------------
  {
    unsigned woff = 0;
    asm volatile("" : "+s"(woff));
    if (wave < 8)
      phase_c(WB + woff, lambA, outseq, b0, Tt - 1, wave, lane, half, nl);
  }
}

extern "C" void kernel_launch(void* const* d_in, const int* in_sizes, int n_in,
                              void* d_out, int out_size, void* d_ws, size_t ws_size,
                              hipStream_t stream) {
  const float* inputs = (const float*)d_in[0];   // [B,T,IN]
  const float* ru     = (const float*)d_in[1];   // [B,T,N]
  const float* J      = (const float*)d_in[2];   // [N,N]
  const float* Bw     = (const float*)d_in[3];   // [N,IN]
  const float* W      = (const float*)d_in[4];   // [OUT,N]

  unsigned short* JB  = (unsigned short*)d_ws;   // 512 KB
  unsigned short* BwB = JB + 512 * 512;          // 128 KB
  unsigned short* WB  = BwB + 512 * 128;         // 128 KB

  float* vt    = (float*)d_out;                              // [B,T+1,N]
  float* zt    = vt + (long)Bb * (Tt + 1) * NN;              // [B,T+1,N]
  float* outsq = zt + (long)Bb * (Tt + 1) * NN;              // [B,T,OUT]

  pack_J <<<(512 * 512) / 256, 256, 0, stream>>>(J, JB);
  pack_Bw<<<(512 * 128) / 256, 256, 0, stream>>>(Bw, BwB);
  pack_W <<<(128 * 512) / 256, 256, 0, stream>>>(W, WB);

  rsnn_kernel<<<Bb / BM, THREADS, 0, stream>>>(inputs, ru, JB, BwB, WB,
                                               vt, zt, outsq);
}
